// GNN_Discriminator_60430189855245
// MI455X (gfx1250) — matrix operand order repaired and checked by reading, not compiled, not run on hardware
//
#include <hip/hip_runtime.h>
#include <hip/hip_bf16.h>
#include <math.h>

typedef __attribute__((ext_vector_type(16))) _Float16 v16h;
typedef __attribute__((ext_vector_type(8)))  _Float16 v8h;
typedef __attribute__((ext_vector_type(8)))  float    v8f;

// ---------------------------------------------------------------------------
// Packed-weight offsets (in halves). Each 32x16 (KxN) tile = 512 halves, laid
// out in per-lane B-fragment order for v_wmma_f32_16x16x32_f16:
//   lane l: column n = l&15, h = l>>4, j<8 -> K=8h+j, j>=8 -> K=16+8h+(j-8).
// Tile index = kc*Nt + nc.
// ---------------------------------------------------------------------------
#define OFF_CUR_W0 0        // 8->128   Kt=1 Nt=8   (4096)
#define OFF_CUR_W1 4096     // 128->256 Kt=4 Nt=16  (32768)
#define OFF_CUR_W2 36864    // 256->128 Kt=8 Nt=8   (32768)
#define OFF_CUR_W3 69632    // 128->64  Kt=4 Nt=4   (8192)
#define OFF_IN_W0  77824
#define OFF_IN_W1  81920
#define OFF_IN_W2  114688
#define OFF_IN_W3  147456
#define OFF_FIN_W0 155648   // 64->128  Kt=2 Nt=8   (8192)
#define OFF_FIN_W1 163840
#define OFF_FIN_W2 196608
#define OFF_FIN_W3 229376   // 128->8   Kt=4 Nt=1   (2048)
#define W_TOTAL_HALVES 231424

// ---------------------------------------------------------------------------
// Weight packing: fp32 [in_f, out_f] row-major -> f16 B fragments, zero-padded
// ---------------------------------------------------------------------------
__global__ void pack_w_kernel(const float* __restrict__ W, _Float16* __restrict__ dst,
                              int in_f, int out_f, int Nt, int total) {
    int idx = blockIdx.x * blockDim.x + threadIdx.x;
    if (idx >= total) return;
    int tile   = idx >> 9;
    int within = idx & 511;
    int lane   = within >> 4;
    int j      = within & 15;
    int kc = tile / Nt;
    int nc = tile - kc * Nt;
    int n     = lane & 15;
    int kbase = (lane >> 4) * 16;
    int K   = kc * 32 + (j < 8 ? (kbase >> 1) + j : 16 + (kbase >> 1) + (j - 8));
    int col = nc * 16 + n;
    float v = (K < in_f && col < out_f) ? W[K * out_f + col] : 0.0f;
    dst[idx] = (_Float16)v;
}

// X padded to 32 cols f16: [0..6]=vertex, [7]=label, [8..31]=0
__global__ void init_x_kernel(const float* __restrict__ vertex, const float* __restrict__ label,
                              _Float16* __restrict__ Xpad, int N) {
    int idx = blockIdx.x * blockDim.x + threadIdx.x;
    if (idx >= N * 32) return;
    int row = idx >> 5;
    int c   = idx & 31;
    float v = (c < 7) ? vertex[row * 7 + c] : (c == 7 ? label[row] : 0.0f);
    Xpad[idx] = (_Float16)v;
}

// ---------------------------------------------------------------------------
// One hidden MLP layer for a 16-row tile, one wave. actIn: LDS [16][Kt*32] f16
// row-major. actOut: LDS [16][outW] f16. ReLU applied.
// ---------------------------------------------------------------------------
__device__ __forceinline__ void mlp_layer(const _Float16* __restrict__ Wf,
                                          const float* __restrict__ bias,
                                          const _Float16* actIn, _Float16* actOut,
                                          int Kt, int Nt, int outW, int lane) {
    const int m    = lane & 15;      // A row / D col index
    const int hsel = lane >> 4;      // half-wave select
    const int n    = m;              // D column
    const int inW  = Kt * 32;
    for (int nc = 0; nc < Nt; ++nc) {
        v8f acc = {};
        for (int kc = 0; kc < Kt; ++kc) {
            const _Float16* rp = actIn + m * inW + kc * 32;
            v8h lo = *(const v8h*)(rp + hsel * 8);
            v8h hi = *(const v8h*)(rp + 16 + hsel * 8);
            v16h a = __builtin_shufflevector(lo, hi, 0,1,2,3,4,5,6,7,8,9,10,11,12,13,14,15);
            v16h b = *(const v16h*)(Wf + (size_t)(kc * Nt + nc) * 512 + lane * 16);
            acc = __builtin_amdgcn_wmma_f32_16x16x32_f16(false, a, false, b,
                                                         (short)0, acc, false, false);
        }
        float bv = bias[nc * 16 + n];
        for (int v = 0; v < 8; ++v) {
            float x = acc[v] + bv;
            x = x > 0.0f ? x : 0.0f;
            actOut[(hsel * 8 + v) * outW + nc * 16 + n] = (_Float16)x;
        }
    }
}

// Final 128->64 layer (linear), write f32 to out[row*64+col]; optional LDS S sum
__device__ __forceinline__ void final64_layer(const _Float16* __restrict__ Wf,
                                              const float* __restrict__ bias,
                                              const _Float16* actIn, float* __restrict__ out,
                                              int tile0, int N, int lane, float* sS) {
    const int m = lane & 15, hsel = lane >> 4, n = m;
    for (int nc = 0; nc < 4; ++nc) {
        v8f acc = {};
        for (int kc = 0; kc < 4; ++kc) {
            const _Float16* rp = actIn + m * 128 + kc * 32;
            v8h lo = *(const v8h*)(rp + hsel * 8);
            v8h hi = *(const v8h*)(rp + 16 + hsel * 8);
            v16h a = __builtin_shufflevector(lo, hi, 0,1,2,3,4,5,6,7,8,9,10,11,12,13,14,15);
            v16h b = *(const v16h*)(Wf + (size_t)(kc * 4 + nc) * 512 + lane * 16);
            acc = __builtin_amdgcn_wmma_f32_16x16x32_f16(false, a, false, b,
                                                         (short)0, acc, false, false);
        }
        float bv = bias[nc * 16 + n];
        float psum = 0.0f;
        for (int v = 0; v < 8; ++v) {
            int row = tile0 + hsel * 8 + v;
            float x = acc[v] + bv;
            if (row < N) {
                out[(size_t)row * 64 + nc * 16 + n] = x;
                psum += x;
            }
        }
        if (sS) atomicAdd(&sS[nc * 16 + n], psum);
    }
}

// ---------------------------------------------------------------------------
// Kernel A: c = cur_mlp(X), u = in_mlp(X), S += sum_rows(u)
// 128 threads = 4 waves, 16 rows/wave -> 64 rows/block
// ---------------------------------------------------------------------------
__global__ __launch_bounds__(128) void gnn_cur_in_kernel(
    const _Float16* __restrict__ Wall,
    const float* cb0, const float* cb1, const float* cb2, const float* cb3,
    const float* ib0, const float* ib1, const float* ib2, const float* ib3,
    const _Float16* __restrict__ Xpad,
    float* __restrict__ Cout, float* __restrict__ Uout,
    float* __restrict__ Sglob, int N) {
    __shared__ _Float16 smX[4 * 16 * 32];
    __shared__ _Float16 smB1[4 * 16 * 128];
    __shared__ _Float16 smB2[4 * 16 * 256];
    __shared__ float sS[64];

    const int tid = threadIdx.x, wave = tid >> 5, lane = tid & 31;
    if (tid < 64) sS[tid] = 0.0f;
    const int tile0 = (blockIdx.x * 4 + wave) * 16;
    _Float16* Xb = smX + wave * 512;
    _Float16* B1 = smB1 + wave * 2048;
    _Float16* B2 = smB2 + wave * 4096;
    const int m = lane & 15, hsel = lane >> 4;

    {   // load 16x32 X tile (clamped rows; tail masked at stores)
        int row = tile0 + m;
        if (row >= N) row = N - 1;
        const v8h* src = (const v8h*)(Xpad + (size_t)row * 32 + hsel * 16);
        v8h* dst = (v8h*)(Xb + m * 32 + hsel * 16);
        dst[0] = src[0];
        dst[1] = src[1];
    }
    __syncthreads();
    // ---- current_update ----
    mlp_layer(Wall + OFF_CUR_W0, cb0, Xb, B1, 1, 8, 128, lane);  __syncthreads();
    mlp_layer(Wall + OFF_CUR_W1, cb1, B1, B2, 4, 16, 256, lane); __syncthreads();
    mlp_layer(Wall + OFF_CUR_W2, cb2, B2, B1, 8, 8, 128, lane);  __syncthreads();
    final64_layer(Wall + OFF_CUR_W3, cb3, B1, Cout, tile0, N, lane, nullptr);
    __syncthreads();
    // ---- in_update ----
    mlp_layer(Wall + OFF_IN_W0, ib0, Xb, B1, 1, 8, 128, lane);   __syncthreads();
    mlp_layer(Wall + OFF_IN_W1, ib1, B1, B2, 4, 16, 256, lane);  __syncthreads();
    mlp_layer(Wall + OFF_IN_W2, ib2, B2, B1, 8, 8, 128, lane);   __syncthreads();
    final64_layer(Wall + OFF_IN_W3, ib3, B1, Uout, tile0, N, lane, sS);
    __syncthreads();
    if (tid < 64) atomicAdd(&Sglob[tid], sS[tid]);
}

// ---------------------------------------------------------------------------
// Kernel B: avg = (c + S - u)/N ; X = fin_mlp(avg) ; (last step) sumX += X
// ---------------------------------------------------------------------------
__global__ __launch_bounds__(128) void gnn_fin_kernel(
    const _Float16* __restrict__ Wall,
    const float* fb0, const float* fb1, const float* fb2, const float* fb3,
    const float* __restrict__ Cin, const float* __restrict__ Uin,
    const float* __restrict__ Sglob,
    _Float16* __restrict__ Xpad,
    float* __restrict__ sumX, int N, int accumulate) {
    __shared__ _Float16 smA[4 * 16 * 64];
    __shared__ _Float16 smB1[4 * 16 * 128];
    __shared__ _Float16 smB2[4 * 16 * 256];
    __shared__ float sS[64];
    __shared__ float sX[8];

    const int tid = threadIdx.x, wave = tid >> 5, lane = tid & 31;
    if (tid < 64) sS[tid] = Sglob[tid];
    if (tid < 8) sX[tid] = 0.0f;
    __syncthreads();

    const int tile0 = (blockIdx.x * 4 + wave) * 16;
    _Float16* Ab = smA + wave * 1024;
    _Float16* B1 = smB1 + wave * 2048;
    _Float16* B2 = smB2 + wave * 4096;
    const int m = lane & 15, hsel = lane >> 4;
    const float invN = 1.0f / (float)N;

    {   // avg tile [16][64] f16
        int row = tile0 + m;
        if (row >= N) row = N - 1;
        const float* cp = Cin + (size_t)row * 64 + hsel * 32;
        const float* up = Uin + (size_t)row * 64 + hsel * 32;
        for (int j = 0; j < 32; ++j) {
            float v = (cp[j] + sS[hsel * 32 + j] - up[j]) * invN;
            Ab[m * 64 + hsel * 32 + j] = (_Float16)v;
        }
    }
    __syncthreads();
    mlp_layer(Wall + OFF_FIN_W0, fb0, Ab, B1, 2, 8, 128, lane);  __syncthreads();
    mlp_layer(Wall + OFF_FIN_W1, fb1, B1, B2, 4, 16, 256, lane); __syncthreads();
    mlp_layer(Wall + OFF_FIN_W2, fb2, B2, B1, 8, 8, 128, lane);  __syncthreads();
    {   // final 128 -> 8 (N padded to 16)
        const int n = lane & 15;
        v8f acc = {};
        for (int kc = 0; kc < 4; ++kc) {
            const _Float16* rp = B1 + m * 128 + kc * 32;
            v8h lo = *(const v8h*)(rp + hsel * 8);
            v8h hi = *(const v8h*)(rp + 16 + hsel * 8);
            v16h a = __builtin_shufflevector(lo, hi, 0,1,2,3,4,5,6,7,8,9,10,11,12,13,14,15);
            v16h b = *(const v16h*)(Wall + OFF_FIN_W3 + (size_t)kc * 512 + lane * 16);
            acc = __builtin_amdgcn_wmma_f32_16x16x32_f16(false, a, false, b,
                                                         (short)0, acc, false, false);
        }
        if (n < 8) {
            float bv = fb3[n];
            float psum = 0.0f;
            for (int v = 0; v < 8; ++v) {
                int row = tile0 + hsel * 8 + v;
                if (row < N) {
                    float x = acc[v] + bv;
                    Xpad[(size_t)row * 32 + n] = (_Float16)x;
                    psum += x;
                }
            }
            if (accumulate) atomicAdd(&sX[n], psum);
        }
    }
    __syncthreads();
    if (accumulate && tid < 8) atomicAdd(&sumX[tid], sX[tid]);
}

// ---------------------------------------------------------------------------
// Classifier: avg_x = sumX/N ; 8->128->256->128->1 ; sigmoid. One block.
// ---------------------------------------------------------------------------
__global__ __launch_bounds__(256) void gnn_cls_kernel(
    const float* __restrict__ sumX,
    const float* W0, const float* b0, const float* W1, const float* b1,
    const float* W2, const float* b2, const float* W3, const float* b3,
    float* __restrict__ out, int N) {
    __shared__ float h0[8];
    __shared__ float h1[128];
    __shared__ float h2[256];
    __shared__ float h3[128];
    __shared__ float red[256];
    const int t = threadIdx.x;
    if (t < 8) h0[t] = sumX[t] / (float)N;
    __syncthreads();
    if (t < 128) {
        float a = b0[t];
        for (int k = 0; k < 8; ++k) a += h0[k] * W0[k * 128 + t];
        h1[t] = a > 0.0f ? a : 0.0f;
    }
    __syncthreads();
    {
        float a = b1[t];
        for (int k = 0; k < 128; ++k) a += h1[k] * W1[k * 256 + t];
        h2[t] = a > 0.0f ? a : 0.0f;
    }
    __syncthreads();
    if (t < 128) {
        float a = b2[t];
        for (int k = 0; k < 256; ++k) a += h2[k] * W2[k * 128 + t];
        h3[t] = a > 0.0f ? a : 0.0f;
    }
    __syncthreads();
    red[t] = (t < 128) ? h3[t] * W3[t] : 0.0f;
    __syncthreads();
    for (int s = 128; s > 0; s >>= 1) {
        if (t < s) red[t] += red[t + s];
        __syncthreads();
    }
    if (t == 0) {
        float z = red[0] + b3[0];
        out[0] = 1.0f / (1.0f + expf(-z));
    }
}

// ---------------------------------------------------------------------------
extern "C" void kernel_launch(void* const* d_in, const int* in_sizes, int n_in,
                              void* d_out, int out_size, void* d_ws, size_t ws_size,
                              hipStream_t stream) {
    const float* vertex = (const float*)d_in[0];
    const float* label  = (const float*)d_in[2];
    const int N = in_sizes[0] / 7;

    char* ws = (char*)d_ws;
    _Float16* Wall = (_Float16*)ws;
    size_t offX   = 524288;                              // packed weights region
    size_t offC   = offX + (((size_t)N * 32 * 2 + 255) & ~(size_t)255);
    size_t offU   = offC + (((size_t)N * 64 * 4 + 255) & ~(size_t)255);
    size_t offS   = offU + (((size_t)N * 64 * 4 + 255) & ~(size_t)255);
    size_t offSum = offS + 256;
    if (ws_size < offSum + 256) return;  // workspace too small; nothing safe to do

    _Float16* Xpad = (_Float16*)(ws + offX);
    float* Cbuf = (float*)(ws + offC);
    float* Ubuf = (float*)(ws + offU);
    float* Sbuf = (float*)(ws + offS);
    float* Xsum = (float*)(ws + offSum);

    // Pack all 12 weight matrices into WMMA B-fragment order (f16)
    struct LT { int src; int off; int inf; int outf; };
    const LT lt[12] = {
        {3,  OFF_CUR_W0,   8, 128}, {5,  OFF_CUR_W1, 128, 256},
        {7,  OFF_CUR_W2, 256, 128}, {9,  OFF_CUR_W3, 128,  64},
        {11, OFF_IN_W0,    8, 128}, {13, OFF_IN_W1,  128, 256},
        {15, OFF_IN_W2,  256, 128}, {17, OFF_IN_W3,  128,  64},
        {19, OFF_FIN_W0,  64, 128}, {21, OFF_FIN_W1, 128, 256},
        {23, OFF_FIN_W2, 256, 128}, {25, OFF_FIN_W3, 128,   8},
    };
    for (int i = 0; i < 12; ++i) {
        int Kt = (lt[i].inf + 31) / 32;
        int Nt = (lt[i].outf + 15) / 16;
        int total = Kt * Nt * 512;
        pack_w_kernel<<<(total + 255) / 256, 256, 0, stream>>>(
            (const float*)d_in[lt[i].src], Wall + lt[i].off, lt[i].inf, lt[i].outf, Nt, total);
    }
    init_x_kernel<<<(N * 32 + 255) / 256, 256, 0, stream>>>(vertex, label, Xpad, N);

    const int blocksA = (N + 63) / 64;
    for (int step = 0; step < 3; ++step) {
        hipMemsetAsync(Sbuf, 0, 64 * sizeof(float), stream);
        gnn_cur_in_kernel<<<blocksA, 128, 0, stream>>>(
            Wall,
            (const float*)d_in[4],  (const float*)d_in[6],
            (const float*)d_in[8],  (const float*)d_in[10],
            (const float*)d_in[12], (const float*)d_in[14],
            (const float*)d_in[16], (const float*)d_in[18],
            Xpad, Cbuf, Ubuf, Sbuf, N);
        const int accum = (step == 2) ? 1 : 0;
        if (accum) hipMemsetAsync(Xsum, 0, 8 * sizeof(float), stream);
        gnn_fin_kernel<<<blocksA, 128, 0, stream>>>(
            Wall,
            (const float*)d_in[20], (const float*)d_in[22],
            (const float*)d_in[24], (const float*)d_in[26],
            Cbuf, Ubuf, Sbuf, Xpad, Xsum, N, accum);
    }
    gnn_cls_kernel<<<1, 256, 0, stream>>>(
        Xsum,
        (const float*)d_in[27], (const float*)d_in[28],
        (const float*)d_in[29], (const float*)d_in[30],
        (const float*)d_in[31], (const float*)d_in[32],
        (const float*)d_in[33], (const float*)d_in[34],
        (float*)d_out, N);
}
